// Net_49031346651801
// MI455X (gfx1250) — matrix operand order repaired and checked by reading, not compiled, not run on hardware
//
#include <hip/hip_runtime.h>
#include <math.h>

// ---- problem constants (from reference) ----
#define N_NODES 170000
#define N_EDGES 2400000
#define F_INP   128
#define HDIM    64
#define C_OUT   40
#define C_PAD   48          // pad classes to 3x16 WMMA column tiles
#define L_LAYERS 8
#define L1       9          // L+1
#define ENTROPY_W 1.0f
#define EPS_T    1e-20f

typedef __attribute__((ext_vector_type(16))) __bf16 v16bf;
typedef __attribute__((ext_vector_type(8)))  float  v8f;

// 16-bit B-matrix 32x16 (KxN) layout: element i<8 holds K=8*kh+i, i>=8 holds
// K=16+8*kh+(i-8) (kh = lane>>4); lane holds column n = lane&15.
__device__ __forceinline__ int kmapB(int i, int kh) { return (i < 8) ? (kh * 8 + i) : (16 + kh * 8 + (i - 8)); }

// ----------------------------------------------------------------------------
__global__ void k_init(float* __restrict__ accum, float* __restrict__ ent_slot) {
  int i = blockIdx.x * blockDim.x + threadIdx.x;
  if (i < N_NODES * HDIM) accum[i] = 0.0f;
  if (i == 0) *ent_slot = 0.0f;
}

// ---- pack one A fragment (16x32 bf16, ISA layout) from two 8-float runs -----
__device__ __forceinline__ v16bf pack_a(const float* ap) {
  float4 f0 = *(const float4*)(ap + 0);
  float4 f1 = *(const float4*)(ap + 4);
  float4 f2 = *(const float4*)(ap + 16);
  float4 f3 = *(const float4*)(ap + 20);
  v16bf a;
  a[0] = (__bf16)f0.x; a[1] = (__bf16)f0.y; a[2]  = (__bf16)f0.z; a[3]  = (__bf16)f0.w;
  a[4] = (__bf16)f1.x; a[5] = (__bf16)f1.y; a[6]  = (__bf16)f1.z; a[7]  = (__bf16)f1.w;
  a[8] = (__bf16)f2.x; a[9] = (__bf16)f2.y; a[10] = (__bf16)f2.z; a[11] = (__bf16)f2.w;
  a[12] = (__bf16)f3.x; a[13] = (__bf16)f3.y; a[14] = (__bf16)f3.z; a[15] = (__bf16)f3.w;
  return a;
}

// ---- lin0: slice0 = relu(x @ W0 + b0), bf16 WMMA, f32 accumulate ------------
// B fragments pre-swizzled in LDS: one contiguous 32B v16bf per (chunk,tile,lane)
__global__ void k_lin0_wmma(const float* __restrict__ x,
                            const float* __restrict__ w,     // [F_INP, HDIM]
                            const float* __restrict__ bias,  // [HDIM]
                            float* __restrict__ slice0) {
  __shared__ __align__(32) __bf16 bfrag[4 * 4 * 32 * 16];  // 16 KB
  for (int idx = threadIdx.x; idx < 4 * 4 * 32 * 16; idx += 256) {
    const int i    = idx & 15;
    const int lane = (idx >> 4) & 31;
    const int j    = (idx >> 9) & 3;
    const int c    = idx >> 11;
    const int k = c * 32 + kmapB(i, lane >> 4);
    const int n = j * 16 + (lane & 15);
    bfrag[idx] = (__bf16)w[k * HDIM + n];
  }
  __syncthreads();

  const int wave = threadIdx.x >> 5, lane = threadIdx.x & 31;
  const int row0 = (blockIdx.x * 8 + wave) * 16;
  const int lm = lane & 15, kh = lane >> 4;
  int arow = row0 + lm;                     // A row for this lane
  if (arow >= N_NODES) arow = N_NODES - 1;  // clamp; EXEC stays all-ones

  v8f acc[4];
#pragma unroll
  for (int j = 0; j < 4; ++j)
#pragma unroll
    for (int e = 0; e < 8; ++e) acc[j][e] = 0.0f;

#pragma unroll
  for (int kc = 0; kc < F_INP; kc += 32) {
    const v16bf a = pack_a(x + (size_t)arow * F_INP + kc + kh * 8);
#pragma unroll
    for (int j = 0; j < 4; ++j) {
      const v16bf b =
          *(const v16bf*)(bfrag + ((((kc >> 5) * 4) + j) * 32 + lane) * 16);
      acc[j] = __builtin_amdgcn_wmma_f32_16x16x32_bf16(
          false, a, false, b, (short)0, acc[j], false, false);
    }
  }
  // D layout: col n = lane&15, row = v + 8*kh. Fused bias + ReLU store.
#pragma unroll
  for (int j = 0; j < 4; ++j) {
    const int n = j * 16 + lm;
    const float bn = bias[n];
#pragma unroll
    for (int v = 0; v < 8; ++v) {
      const int r = row0 + v + 8 * kh;
      if (r < N_NODES) slice0[(size_t)r * HDIM + n] = fmaxf(acc[j][v] + bn, 0.0f);
    }
  }
}

// ---- propagation: accum[dst] += w * x[src], 16 threads/edge, float4 ---------
__global__ void k_scatter(const int* __restrict__ esrc, const int* __restrict__ edst,
                          const float* __restrict__ ew, const float* __restrict__ xin,
                          float* __restrict__ accum) {
  long long tid = (long long)blockIdx.x * blockDim.x + threadIdx.x;
  int e = (int)(tid >> 4);
  if (e >= N_EDGES) return;
  int part = (int)(tid & 15);
  if (part == 0) {  // stream edge arrays ahead of the wave
    int pe = e + 8192;
    if (pe < N_EDGES) {
      __builtin_prefetch(&esrc[pe], 0, 1);
      __builtin_prefetch(&edst[pe], 0, 1);
      __builtin_prefetch(&ew[pe], 0, 1);
    }
  }
  const int s = esrc[e], d = edst[e];
  const float w = ew[e];
  const float4 v = *(const float4*)(xin + (size_t)s * HDIM + part * 4);
  float* dp = accum + (size_t)d * HDIM + part * 4;
  atomicAdd(dp + 0, w * v.x);
  atomicAdd(dp + 1, w * v.y);
  atomicAdd(dp + 2, w * v.z);
  atomicAdd(dp + 3, w * v.w);
}

// ---- next slice = relu(accum); re-zero accum for next layer -----------------
__global__ void k_relu_advance(float* __restrict__ accum, float* __restrict__ nxt) {
  int i = blockIdx.x * blockDim.x + threadIdx.x;
  if (i < N_NODES * HDIM) {
    float v = accum[i];
    accum[i] = 0.0f;
    nxt[i] = fmaxf(v, 0.0f);
  }
}

// ---- per-node epilogue: meta dots, stick-breaking, entropy, gumbel, mix -----
__global__ void k_meta_mix(const float* __restrict__ xmat,   // [L1][N][H]
                           const float* __restrict__ meta_w, // [H]
                           const float* __restrict__ meta_b, // [1]
                           const float* __restrict__ gumbel, // [N][L1]
                           float* __restrict__ mixed,        // [N][H]
                           float* __restrict__ ent_slot) {
  __shared__ float ent_s;
  if (threadIdx.x == 0) ent_s = 0.0f;
  __syncthreads();

  const int wave = threadIdx.x >> 5, lane = threadIdx.x & 31;
  int n = blockIdx.x * 8 + wave;
  const bool valid = (n < N_NODES);
  if (!valid) n = N_NODES - 1;  // clamp; stay convergent for the barriers

  const float mw0 = meta_w[lane], mw1 = meta_w[lane + 32];
  const float mb = meta_b[0];
  float x0[L1], x1[L1], r[L1];
#pragma unroll
  for (int l = 0; l < L1; ++l) {
    const float* p = xmat + (size_t)l * N_NODES * HDIM + (size_t)n * HDIM;
    x0[l] = p[lane];
    x1[l] = p[lane + 32];
    float s = x0[l] * mw0 + x1[l] * mw1;
#pragma unroll
    for (int m = 16; m >= 1; m >>= 1) s += __shfl_xor(s, m, 32);
    r[l] = s + mb;  // all lanes hold the full dot
  }
  // stick-breaking: t[l] = sig(r[l]) * prod_{j<l}(1-sig(r[j])); t[L]=cumprod
  float t[L1];
  float cp = 1.0f;
#pragma unroll
  for (int l = 0; l < L1; ++l) {
    float s = 1.0f / (1.0f + __expf(-r[l]));
    if (l == L1 - 1) {
      t[l] = cp;
    } else {
      t[l] = s * cp;
      cp *= (1.0f - s);
    }
  }
  float ent = 0.0f;
#pragma unroll
  for (int l = 0; l < L1; ++l) ent -= t[l] * __logf(t[l] + EPS_T);
  // block-level reduction: LDS atomic per wave, one global atomic per block
  if (lane == 0 && valid) atomicAdd(&ent_s, ent * (ENTROPY_W / (float)N_NODES));

  // gumbel softmax over log(t+eps)+g
  float z[L1], zmax = -1e30f;
#pragma unroll
  for (int l = 0; l < L1; ++l) {
    z[l] = __logf(t[l] + EPS_T) + gumbel[(size_t)n * L1 + l];
    zmax = fmaxf(zmax, z[l]);
  }
  float se = 0.0f;
#pragma unroll
  for (int l = 0; l < L1; ++l) {
    z[l] = __expf(z[l] - zmax);
    se += z[l];
  }
  const float inv = 1.0f / se;
  float o0 = 0.0f, o1 = 0.0f;
#pragma unroll
  for (int l = 0; l < L1; ++l) {
    const float tm = z[l] * inv;
    o0 += tm * x0[l];
    o1 += tm * x1[l];
  }
  if (valid) {
    mixed[(size_t)n * HDIM + lane] = o0;
    mixed[(size_t)n * HDIM + lane + 32] = o1;
  }
  __syncthreads();
  if (threadIdx.x == 0) atomicAdd(ent_slot, ent_s);
}

// ---- lin1 + log_softmax: bf16 WMMA, row reductions via half-wave shuffles ---
__global__ void k_lin1_wmma_lsm(const float* __restrict__ mixed,  // [N][H]
                                const float* __restrict__ w,      // [H][C_OUT]
                                const float* __restrict__ bias,   // [C_OUT]
                                float* __restrict__ out) {        // [N][C_OUT]
  __shared__ __align__(32) __bf16 bfrag[2 * 3 * 32 * 16];  // 6 KB, pre-swizzled
  for (int idx = threadIdx.x; idx < 2 * 3 * 32 * 16; idx += 256) {
    const int i    = idx & 15;
    const int lane = (idx >> 4) & 31;
    const int rest = idx >> 9;  // 0..5
    const int j = rest % 3;
    const int c = rest / 3;
    const int k = c * 32 + kmapB(i, lane >> 4);
    const int n = j * 16 + (lane & 15);
    bfrag[idx] = (__bf16)((n < C_OUT) ? w[k * C_OUT + n] : 0.0f);
  }
  __syncthreads();

  const int wave = threadIdx.x >> 5, lane = threadIdx.x & 31;
  const int row0 = (blockIdx.x * 8 + wave) * 16;
  const int lm = lane & 15, kh = lane >> 4;
  int arow = row0 + lm;
  if (arow >= N_NODES) arow = N_NODES - 1;

  v8f acc[3];
#pragma unroll
  for (int j = 0; j < 3; ++j)
#pragma unroll
    for (int e = 0; e < 8; ++e) acc[j][e] = 0.0f;

#pragma unroll
  for (int kc = 0; kc < HDIM; kc += 32) {
    const v16bf a = pack_a(mixed + (size_t)arow * HDIM + kc + kh * 8);
#pragma unroll
    for (int j = 0; j < 3; ++j) {
      const v16bf b =
          *(const v16bf*)(bfrag + ((((kc >> 5) * 3) + j) * 32 + lane) * 16);
      acc[j] = __builtin_amdgcn_wmma_f32_16x16x32_bf16(
          false, a, false, b, (short)0, acc[j], false, false);
    }
  }

  // logits with bias; pad cols forced to -1e30 so they vanish in softmax
  float Lg[3][8];
#pragma unroll
  for (int j = 0; j < 3; ++j) {
    const int n = j * 16 + lm;
    const float bn = (n < C_OUT) ? bias[n] : 0.0f;
#pragma unroll
    for (int v = 0; v < 8; ++v)
      Lg[j][v] = (n < C_OUT) ? (acc[j][v] + bn) : -1e30f;
  }
  // row m = v + 8*kh lives across one 16-lane half: reduce with xor 8..1
  float rmax[8], rsum[8];
#pragma unroll
  for (int v = 0; v < 8; ++v) {
    float p = fmaxf(fmaxf(Lg[0][v], Lg[1][v]), Lg[2][v]);
#pragma unroll
    for (int m = 8; m >= 1; m >>= 1) p = fmaxf(p, __shfl_xor(p, m, 32));
    rmax[v] = p;
  }
#pragma unroll
  for (int v = 0; v < 8; ++v) {
    float q = __expf(Lg[0][v] - rmax[v]) + __expf(Lg[1][v] - rmax[v]) +
              __expf(Lg[2][v] - rmax[v]);
#pragma unroll
    for (int m = 8; m >= 1; m >>= 1) q += __shfl_xor(q, m, 32);
    rsum[v] = q;
  }
#pragma unroll
  for (int j = 0; j < 3; ++j) {
    const int n = j * 16 + lm;
    if (n < C_OUT) {
#pragma unroll
      for (int v = 0; v < 8; ++v) {
        const int r = row0 + v + 8 * kh;
        if (r < N_NODES)
          out[(size_t)r * C_OUT + n] = Lg[j][v] - rmax[v] - __logf(rsum[v]);
      }
    }
  }
}

// ----------------------------------------------------------------------------
extern "C" void kernel_launch(void* const* d_in, const int* in_sizes, int n_in,
                              void* d_out, int out_size, void* d_ws, size_t ws_size,
                              hipStream_t stream) {
  const float* x       = (const float*)d_in[0];
  const float* ew      = (const float*)d_in[1];
  const float* gumbel  = (const float*)d_in[2];
  const float* lin0_w  = (const float*)d_in[3];
  const float* lin0_b  = (const float*)d_in[4];
  const float* meta_w  = (const float*)d_in[5];
  const float* meta_b  = (const float*)d_in[6];
  const float* lin1_w  = (const float*)d_in[7];
  const float* lin1_b  = (const float*)d_in[8];
  const int*   esrc    = (const int*)d_in[9];
  const int*   edst    = (const int*)d_in[10];

  float* out      = (float*)d_out;
  float* ent_slot = out + (size_t)N_NODES * C_OUT;  // scalar ent_loss at tail

  float* xmat  = (float*)d_ws;                        // [9][N][H]
  float* accum = xmat + (size_t)L1 * N_NODES * HDIM;  // [N][H]; reused as "mixed"

  const int nh = N_NODES * HDIM;
  k_init<<<(nh + 255) / 256, 256, 0, stream>>>(accum, ent_slot);

  k_lin0_wmma<<<(N_NODES + 127) / 128, 256, 0, stream>>>(x, lin0_w, lin0_b, xmat);

  const long long sthreads = (long long)N_EDGES * 16;
  const int sblocks = (int)((sthreads + 255) / 256);
  for (int l = 0; l < L_LAYERS; ++l) {
    const float* xin = xmat + (size_t)l * nh;
    float* nxt = xmat + (size_t)(l + 1) * nh;
    k_scatter<<<sblocks, 256, 0, stream>>>(esrc, edst, ew, xin, accum);
    k_relu_advance<<<(nh + 255) / 256, 256, 0, stream>>>(accum, nxt);
  }

  k_meta_mix<<<(N_NODES + 7) / 8, 256, 0, stream>>>(xmat, meta_w, meta_b, gumbel,
                                                    accum /*mixed*/, ent_slot);

  k_lin1_wmma_lsm<<<(N_NODES + 127) / 128, 256, 0, stream>>>(accum, lin1_w, lin1_b, out);
}
